// VectorQuantizer_46901042873038
// MI455X (gfx1250) — compile-verified, matrix operand order checked by reading, not run on hardware
//
#include <hip/hip_runtime.h>
#include <hip/hip_bf16.h>

typedef __attribute__((ext_vector_type(16))) __bf16 v16bf;
typedef __attribute__((ext_vector_type(8)))  __bf16 v8bf;
typedef __attribute__((ext_vector_type(8)))  float  v8f;

#define N_ROWS 131072
#define DIM    64
#define K_CODES 4096
#define CHUNK  256   // codes staged in LDS per iteration
#define RT     2     // 16-row tiles per wave (32 rows/wave)

// Rows per block: 8 waves * RT*16 = 256.

__global__ __launch_bounds__(256) void vq_argmin_kernel(
    const float* __restrict__ z, const float* __restrict__ emb,
    float* __restrict__ out)
{
    // B staged in WMMA bf16 lane layout: [tile16][kchunk][hi/lo][lane][16 bf16]
    __shared__ __attribute__((aligned(32))) __bf16 s_b[16][2][2][32][16];
    __shared__ float s_esq[CHUNK];

    const int tid  = threadIdx.x;
    const int lane = tid & 31;
    const int wv   = tid >> 5;
    const int m    = lane & 15;   // row/col within 16
    const int h    = lane >> 4;   // lane-half selects K sub-ranges

    // ---- Build A (z rows) once, in WMMA bf16 A-layout, hi/lo compensated ----
    // A elems 0..7  = dims 32c+8h .. +7 ; elems 8..15 = dims 32c+16+8h .. +7
    v16bf a_hi[RT][2], a_lo[RT][2];
    #pragma unroll
    for (int rt = 0; rt < RT; ++rt) {
        const int row = blockIdx.x * 256 + wv * (RT * 16) + rt * 16 + m;
        const float4* zrow = (const float4*)(z + (size_t)row * DIM);
        #pragma unroll
        for (int c = 0; c < 2; ++c) {
            #pragma unroll
            for (int g = 0; g < 2; ++g) {
                const int dbase = 32 * c + 16 * g + 8 * h;
                float4 f0 = zrow[dbase >> 2];
                float4 f1 = zrow[(dbase >> 2) + 1];
                float vals[8] = {f0.x, f0.y, f0.z, f0.w, f1.x, f1.y, f1.z, f1.w};
                #pragma unroll
                for (int i = 0; i < 8; ++i) {
                    __bf16 hb = (__bf16)vals[i];
                    a_hi[rt][c][8 * g + i] = hb;
                    a_lo[rt][c][8 * g + i] = (__bf16)(vals[i] - (float)hb);
                }
            }
        }
    }

    float best[RT][8];
    int   bidx[RT][8];
    #pragma unroll
    for (int rt = 0; rt < RT; ++rt)
        #pragma unroll
        for (int v = 0; v < 8; ++v) { best[rt][v] = 3.4028235e38f; bidx[rt][v] = 0; }

    for (int kb = 0; kb < K_CODES; kb += CHUNK) {
        __syncthreads();   // protect LDS from previous iteration's readers
        {   // ---- stage: one code row per thread -> bf16 hi/lo in B layout ----
            const int r    = tid;
            const int tt   = r >> 4;
            const int n    = r & 15;
            const float4* erow = (const float4*)(emb + (size_t)(kb + r) * DIM);
            float esq = 0.f;
            #pragma unroll
            for (int c = 0; c < 2; ++c)
            #pragma unroll
            for (int g = 0; g < 2; ++g)
            #pragma unroll
            for (int hh = 0; hh < 2; ++hh) {
                const int dbase = 32 * c + 16 * g + 8 * hh;
                float4 f0 = erow[dbase >> 2];
                float4 f1 = erow[(dbase >> 2) + 1];
                float vals[8] = {f0.x, f0.y, f0.z, f0.w, f1.x, f1.y, f1.z, f1.w};
                v8bf hv, lv;
                #pragma unroll
                for (int i = 0; i < 8; ++i) {
                    float vv = vals[i];
                    esq += vv * vv;
                    __bf16 hb = (__bf16)vv;
                    hv[i] = hb;
                    lv[i] = (__bf16)(vv - (float)hb);
                }
                *(v8bf*)&s_b[tt][c][0][n + 16 * hh][8 * g] = hv;
                *(v8bf*)&s_b[tt][c][1][n + 16 * hh][8 * g] = lv;
            }
            s_esq[r] = esq;
        }
        __syncthreads();

        // ---- WMMA sweep over 16 code tiles in this chunk ----
        #pragma unroll 4
        for (int tt = 0; tt < 16; ++tt) {
            v16bf bh0 = *(const v16bf*)&s_b[tt][0][0][lane][0];
            v16bf bh1 = *(const v16bf*)&s_b[tt][1][0][lane][0];
            v16bf bl0 = *(const v16bf*)&s_b[tt][0][1][lane][0];
            v16bf bl1 = *(const v16bf*)&s_b[tt][1][1][lane][0];
            const float en   = s_esq[tt * 16 + m];
            const int   code = kb + tt * 16 + m;

            // Two independent accumulation chains, interleaved step-major so
            // each dependent WMMA gets a free slot (hides TRANS32 delay) and
            // the WMMA->VALU hazard is paid once for both row-tiles.
            v8f acc[RT];
            #pragma unroll
            for (int rt = 0; rt < RT; ++rt)
                acc[rt] = (v8f){0.f, 0.f, 0.f, 0.f, 0.f, 0.f, 0.f, 0.f};
            #pragma unroll
            for (int rt = 0; rt < RT; ++rt)
                acc[rt] = __builtin_amdgcn_wmma_f32_16x16x32_bf16(false, a_hi[rt][0], false, bh0, (short)0, acc[rt], false, false);
            #pragma unroll
            for (int rt = 0; rt < RT; ++rt)
                acc[rt] = __builtin_amdgcn_wmma_f32_16x16x32_bf16(false, a_hi[rt][1], false, bh1, (short)0, acc[rt], false, false);
            #pragma unroll
            for (int rt = 0; rt < RT; ++rt)
                acc[rt] = __builtin_amdgcn_wmma_f32_16x16x32_bf16(false, a_lo[rt][0], false, bh0, (short)0, acc[rt], false, false);
            #pragma unroll
            for (int rt = 0; rt < RT; ++rt)
                acc[rt] = __builtin_amdgcn_wmma_f32_16x16x32_bf16(false, a_lo[rt][1], false, bh1, (short)0, acc[rt], false, false);
            #pragma unroll
            for (int rt = 0; rt < RT; ++rt)
                acc[rt] = __builtin_amdgcn_wmma_f32_16x16x32_bf16(false, a_hi[rt][0], false, bl0, (short)0, acc[rt], false, false);
            #pragma unroll
            for (int rt = 0; rt < RT; ++rt)
                acc[rt] = __builtin_amdgcn_wmma_f32_16x16x32_bf16(false, a_hi[rt][1], false, bl1, (short)0, acc[rt], false, false);

            // score = ||e||^2 - 2*dot ; keep running min (first index wins ties)
            #pragma unroll
            for (int rt = 0; rt < RT; ++rt)
                #pragma unroll
                for (int v = 0; v < 8; ++v) {
                    float sc = fmaf(-2.f, acc[rt][v], en);
                    if (sc < best[rt][v]) { best[rt][v] = sc; bidx[rt][v] = code; }
                }
        }
    }

    // ---- reduce winner across the 16 N-lanes of each half-wave ----
    #pragma unroll
    for (int off = 1; off < 16; off <<= 1) {
        #pragma unroll
        for (int rt = 0; rt < RT; ++rt)
        #pragma unroll
        for (int v = 0; v < 8; ++v) {
            float os = __shfl_xor(best[rt][v], off, 32);
            int   oi = __shfl_xor(bidx[rt][v], off, 32);
            if (os < best[rt][v] || (os == best[rt][v] && oi < bidx[rt][v])) {
                best[rt][v] = os; bidx[rt][v] = oi;
            }
        }
    }
    if (m == 0) {   // lanes 0 and 16 each own 8 rows per row-tile (M = v + 8h)
        #pragma unroll
        for (int rt = 0; rt < RT; ++rt)
        #pragma unroll
        for (int v = 0; v < 8; ++v) {
            const int grow = blockIdx.x * 256 + wv * (RT * 16) + rt * 16 + v + 8 * h;
            out[(size_t)N_ROWS * DIM + grow] = (float)bidx[rt][v];
        }
    }
}

// ---- gather zq (exact fp32) + per-block loss partials ----
__global__ __launch_bounds__(256) void vq_gather_kernel(
    const float* __restrict__ z, const float* __restrict__ emb,
    float* __restrict__ out, float* __restrict__ partial)
{
    const int row = blockIdx.x * 256 + threadIdx.x;
    const int idx = (int)out[(size_t)N_ROWS * DIM + row];
    const float4* e4 = (const float4*)(emb + (size_t)idx * DIM);
    const float4* z4 = (const float4*)(z   + (size_t)row * DIM);
    float4*       o4 = (float4*)(out + (size_t)row * DIM);
    float s = 0.f;
    #pragma unroll
    for (int j = 0; j < DIM / 4; ++j) {
        float4 e = e4[j], zz = z4[j];
        o4[j] = e;  // zq_st == zq in forward value
        float dx = e.x - zz.x, dy = e.y - zz.y, dz = e.z - zz.z, dw = e.w - zz.w;
        s += dx * dx + dy * dy + dz * dz + dw * dw;
    }
    __shared__ float red[256];
    red[threadIdx.x] = s;
    __syncthreads();
    for (int o = 128; o > 0; o >>= 1) {
        if (threadIdx.x < o) red[threadIdx.x] += red[threadIdx.x + o];
        __syncthreads();
    }
    if (threadIdx.x == 0) partial[blockIdx.x] = red[0];
}

__global__ __launch_bounds__(256) void vq_loss_kernel(
    const float* __restrict__ partial, float* __restrict__ out)
{
    __shared__ float red[256];
    red[threadIdx.x] = partial[threadIdx.x] + partial[threadIdx.x + 256];
    __syncthreads();
    for (int o = 128; o > 0; o >>= 1) {
        if (threadIdx.x < o) red[threadIdx.x] += red[threadIdx.x + o];
        __syncthreads();
    }
    if (threadIdx.x == 0)
        out[(size_t)N_ROWS * DIM + N_ROWS] =
            1.5f * red[0] / (float)((size_t)N_ROWS * DIM);  // (BETA + 1) * mean
}

extern "C" void kernel_launch(void* const* d_in, const int* in_sizes, int n_in,
                              void* d_out, int out_size, void* d_ws, size_t ws_size,
                              hipStream_t stream)
{
    (void)in_sizes; (void)n_in; (void)out_size; (void)ws_size;
    const float* z   = (const float*)d_in[0];
    const float* emb = (const float*)d_in[1];
    float* out = (float*)d_out;
    float* partial = (float*)d_ws;     // 512 floats of scratch

    vq_argmin_kernel<<<N_ROWS / 256, 256, 0, stream>>>(z, emb, out);
    vq_gather_kernel<<<N_ROWS / 256, 256, 0, stream>>>(z, emb, out, partial);
    vq_loss_kernel<<<1, 256, 0, stream>>>(partial, out);
}
